// KV_Extension_54279796687481
// MI455X (gfx1250) — compile-verified, matrix operand order checked by reading, not run on hardware
//
#include <hip/hip_runtime.h>
#include <math.h>

// ---- problem constants (match reference) ----
#define BB   16       // batch (HG*WG)
#define CC   768      // feature dim
#define SS   256      // per-grid spatial (16*16)
#define BSZ  4096     // BB*SS
#define NH   12
#define DHD  64
#define CRIT 0.7f
#define TEMP 0.07f
#define CUTV 0.2f

typedef _Float16 half_t;
typedef __attribute__((ext_vector_type(16))) _Float16 v16h;
typedef __attribute__((ext_vector_type(8)))  float    v8f;

// ------------------------------------------------------------------
// Fragment loaders (CDNA5 16-bit A/B 16x32 layout, wave32):
// lanes 0-15 : elems 0..7 = K 0..7,   elems 8..15 = K 16..23   (M/N = lane)
// lanes16-31 : elems 0..7 = K 8..15,  elems 8..15 = K 24..31   (M/N = lane-16)
// ------------------------------------------------------------------
__device__ __forceinline__ v16h load_frag_h(const half_t* __restrict__ base,
                                            int ldk, int row, int k0, int lane) {
  const half_t* p = base + (size_t)row * ldk + k0 + ((lane & 16) ? 8 : 0);
  v16h f;
#pragma unroll
  for (int j = 0; j < 8; ++j) { f[j] = p[j]; f[j + 8] = p[j + 16]; }
  return f;
}

// A-fragment built on the fly from f32 attn thresholded at CRIT (0/1 mask)
__device__ __forceinline__ v16h load_frag_mask(const float* __restrict__ base,
                                               int ldk, int row, int k0, int lane) {
  const float* p = base + (size_t)row * ldk + k0 + ((lane & 16) ? 8 : 0);
  v16h f;
#pragma unroll
  for (int j = 0; j < 8; ++j) {
    f[j]     = (p[j]      > CRIT) ? (half_t)1.0f : (half_t)0.0f;
    f[j + 8] = (p[j + 16] > CRIT) ? (half_t)1.0f : (half_t)0.0f;
  }
  return f;
}

__device__ __forceinline__ void store_tile(float* __restrict__ C, int ldc,
                                           int tm, int tn, int lane, const v8f& acc) {
  int nl = lane & 15;
  int mb = (lane & 16) ? 8 : 0;
#pragma unroll
  for (int r = 0; r < 8; ++r)
    C[(size_t)(tm + mb + r) * ldc + tn + nl] = acc[r];
}

#define WMMA_F16(a, b, c) \
  __builtin_amdgcn_wmma_f32_16x16x32_f16(false, (a), false, (b), (short)0, (c), false, false)

// ------------------------------------------------------------------
// K0: key[n][c] = normalize_c( ex[(b*CC+c)*SS + s] ),  n = b*256+s
// ------------------------------------------------------------------
__global__ void k_build_key(const float* __restrict__ ex,
                            half_t* __restrict__ key_h,
                            half_t* __restrict__ keyT_h) {
  int n = blockIdx.x;           // 0..4095
  int b = n >> 8, s = n & 255;
  int tid = threadIdx.x;        // 256 threads
  __shared__ float red[256];
  float vals[3]; float ss = 0.f;
#pragma unroll
  for (int i = 0; i < 3; ++i) {
    int c = tid + i * 256;
    float v = ex[((size_t)b * CC + c) * SS + s];
    vals[i] = v; ss += v * v;
  }
  red[tid] = ss; __syncthreads();
  for (int off = 128; off > 0; off >>= 1) {
    if (tid < off) red[tid] += red[tid + off];
    __syncthreads();
  }
  float inv = 1.0f / fmaxf(sqrtf(red[0]), 1e-6f);
#pragma unroll
  for (int i = 0; i < 3; ++i) {
    int c = tid + i * 256;
    half_t hv = (half_t)(vals[i] * inv);
    key_h[(size_t)n * CC + c]  = hv;
    keyT_h[(size_t)c * BSZ + n] = hv;
  }
}

// ------------------------------------------------------------------
// K1: VT[c][n] = v_ext[b][h][s][d],  c = h*64+d, n = b*256+s
//     (16x16 -> 16x16 bilinear resize in the reference is identity)
// ------------------------------------------------------------------
__global__ void k_build_v(const float* __restrict__ vext, half_t* __restrict__ VT) {
  size_t i = (size_t)blockIdx.x * 256 + threadIdx.x;  // over 16*12*256*64
  int d = (int)(i & 63);
  size_t r = i >> 6;
  int s = (int)(r & 255); r >>= 8;
  int h = (int)(r % NH);
  int b = (int)(r / NH);
  int c = h * DHD + d;
  int n = b * SS + s;
  VT[(size_t)c * BSZ + n] = (half_t)vext[i];
}

// ------------------------------------------------------------------
// GEMM: D(MxN, f32) = A(MxK f16 rows) * B(NxK f16 rows)^T
// 32x32 tile per wave (2x2 register blocking), 4 waves per block
// ------------------------------------------------------------------
__global__ void k_gemm_nt(const half_t* __restrict__ A, const half_t* __restrict__ B,
                          float* __restrict__ C, int M, int N, int K) {
  int lane = threadIdx.x & 31;
  int wid  = threadIdx.x >> 5;
  int t = blockIdx.x * 4 + wid;
  int ntiles = N >> 5;
  int tm = (t / ntiles) << 5;
  int tn = (t % ntiles) << 5;
  if (tm >= M) return;                       // wave-uniform
  int am0 = tm + (lane & 15), am1 = am0 + 16;
  int bn0 = tn + (lane & 15), bn1 = bn0 + 16;
  v8f acc00 = {}, acc01 = {}, acc10 = {}, acc11 = {};
  for (int k0 = 0; k0 < K; k0 += 32) {
    v16h a0 = load_frag_h(A, K, am0, k0, lane);
    v16h a1 = load_frag_h(A, K, am1, k0, lane);
    v16h b0 = load_frag_h(B, K, bn0, k0, lane);
    v16h b1 = load_frag_h(B, K, bn1, k0, lane);
    acc00 = WMMA_F16(a0, b0, acc00);
    acc01 = WMMA_F16(a0, b1, acc01);
    acc10 = WMMA_F16(a1, b0, acc10);
    acc11 = WMMA_F16(a1, b1, acc11);
  }
  store_tile(C, N, tm,      tn,      lane, acc00);
  store_tile(C, N, tm,      tn + 16, lane, acc01);
  store_tile(C, N, tm + 16, tn,      lane, acc10);
  store_tile(C, N, tm + 16, tn + 16, lane, acc11);
}

// ------------------------------------------------------------------
// Masked GEMM: proxy32(MxN) = (Attn > CRIT)(MxK) * key(KxN); B given as keyT (NxK)
// ------------------------------------------------------------------
__global__ void k_gemm_mask(const float* __restrict__ Attn, const half_t* __restrict__ KT,
                            float* __restrict__ P32, int M, int N, int K) {
  int lane = threadIdx.x & 31;
  int wid  = threadIdx.x >> 5;
  int t = blockIdx.x * 4 + wid;
  int ntiles = N >> 5;
  int tm = (t / ntiles) << 5;
  int tn = (t % ntiles) << 5;
  if (tm >= M) return;
  int am0 = tm + (lane & 15), am1 = am0 + 16;
  int bn0 = tn + (lane & 15), bn1 = bn0 + 16;
  v8f acc00 = {}, acc01 = {}, acc10 = {}, acc11 = {};
  for (int k0 = 0; k0 < K; k0 += 32) {
    v16h a0 = load_frag_mask(Attn, K, am0, k0, lane);
    v16h a1 = load_frag_mask(Attn, K, am1, k0, lane);
    v16h b0 = load_frag_h(KT, K, bn0, k0, lane);
    v16h b1 = load_frag_h(KT, K, bn1, k0, lane);
    acc00 = WMMA_F16(a0, b0, acc00);
    acc01 = WMMA_F16(a0, b1, acc01);
    acc10 = WMMA_F16(a1, b0, acc10);
    acc11 = WMMA_F16(a1, b1, acc11);
  }
  store_tile(P32, N, tm,      tn,      lane, acc00);
  store_tile(P32, N, tm,      tn + 16, lane, acc01);
  store_tile(P32, N, tm + 16, tn,      lane, acc10);
  store_tile(P32, N, tm + 16, tn + 16, lane, acc11);
}

// ------------------------------------------------------------------
// Output GEMM: out = P(BSZ x BSZ f16) * V(BSZ x CC); VT given (CC x BSZ),
// rows written permuted: row m=b*256+s -> out[(s*16+b)*768 + c]
// ------------------------------------------------------------------
__device__ __forceinline__ void store_tile_perm(float* __restrict__ out,
                                                int tm, int tn, int lane, const v8f& acc) {
  int nl = lane & 15;
  int mb = (lane & 16) ? 8 : 0;
#pragma unroll
  for (int r = 0; r < 8; ++r) {
    int m = tm + mb + r;
    int b = m >> 8, s = m & 255;
    out[((size_t)s * BB + b) * CC + tn + nl] = acc[r];
  }
}

__global__ void k_gemm_out(const half_t* __restrict__ P, const half_t* __restrict__ VT,
                           float* __restrict__ out) {
  const int M = BSZ, N = CC, K = BSZ;
  int lane = threadIdx.x & 31;
  int wid  = threadIdx.x >> 5;
  int t = blockIdx.x * 4 + wid;
  int ntiles = N >> 5;
  int tm = (t / ntiles) << 5;
  int tn = (t % ntiles) << 5;
  if (tm >= M) return;
  int am0 = tm + (lane & 15), am1 = am0 + 16;
  int bn0 = tn + (lane & 15), bn1 = bn0 + 16;
  v8f acc00 = {}, acc01 = {}, acc10 = {}, acc11 = {};
  for (int k0 = 0; k0 < K; k0 += 32) {
    v16h a0 = load_frag_h(P,  K, am0, k0, lane);
    v16h a1 = load_frag_h(P,  K, am1, k0, lane);
    v16h b0 = load_frag_h(VT, K, bn0, k0, lane);
    v16h b1 = load_frag_h(VT, K, bn1, k0, lane);
    acc00 = WMMA_F16(a0, b0, acc00);
    acc01 = WMMA_F16(a0, b1, acc01);
    acc10 = WMMA_F16(a1, b0, acc10);
    acc11 = WMMA_F16(a1, b1, acc11);
  }
  store_tile_perm(out, tm,      tn,      lane, acc00);
  store_tile_perm(out, tm,      tn + 16, lane, acc01);
  store_tile_perm(out, tm + 16, tn,      lane, acc10);
  store_tile_perm(out, tm + 16, tn + 16, lane, acc11);
}

// ------------------------------------------------------------------
// row count of (attn > CRIT)
// ------------------------------------------------------------------
__global__ void k_count(const float* __restrict__ Attn, float* __restrict__ cnt) {
  int n = blockIdx.x, tid = threadIdx.x;
  __shared__ float red[256];
  const float* row = Attn + (size_t)n * BSZ;
  float c = 0.f;
  for (int i = tid; i < BSZ; i += 256) c += (row[i] > CRIT) ? 1.f : 0.f;
  red[tid] = c; __syncthreads();
  for (int off = 128; off > 0; off >>= 1) {
    if (tid < off) red[tid] += red[tid + off];
    __syncthreads();
  }
  if (tid == 0) cnt[n] = red[0];
}

// ------------------------------------------------------------------
// proxy_h = normalize(proxy32 row), empty rows fall back to key row
// ------------------------------------------------------------------
__global__ void k_norm_proxy(const float* __restrict__ P32, const half_t* __restrict__ key_h,
                             const float* __restrict__ cnt, half_t* __restrict__ proxy_h) {
  int n = blockIdx.x, tid = threadIdx.x;
  bool empty = (cnt[n] == 0.0f);
  __shared__ float red[256];
  float vals[3]; float ss = 0.f;
#pragma unroll
  for (int i = 0; i < 3; ++i) {
    int c = tid + i * 256;
    float v = empty ? (float)key_h[(size_t)n * CC + c] : P32[(size_t)n * CC + c];
    vals[i] = v; ss += v * v;
  }
  red[tid] = ss; __syncthreads();
  for (int off = 128; off > 0; off >>= 1) {
    if (tid < off) red[tid] += red[tid + off];
    __syncthreads();
  }
  float inv = 1.0f / fmaxf(sqrtf(red[0]), 1e-6f);
#pragma unroll
  for (int i = 0; i < 3; ++i) {
    int c = tid + i * 256;
    proxy_h[(size_t)n * CC + c] = (half_t)(vals[i] * inv);
  }
}

// ------------------------------------------------------------------
// row-wise: affine (beta/gamma), dynamic cut, softmax/T -> f16 P
// ------------------------------------------------------------------
__global__ void k_softmax(const float* __restrict__ Attn, const float* __restrict__ cnt,
                          half_t* __restrict__ P) {
  int n = blockIdx.x, tid = threadIdx.x;
  const float* row = Attn + (size_t)n * BSZ;
  __shared__ float red[256];
  float v[16]; float sum = 0.f;
#pragma unroll
  for (int i = 0; i < 16; ++i) { v[i] = row[i * 256 + tid]; sum += v[i]; }

  red[tid] = sum; __syncthreads();
  for (int off = 128; off > 0; off >>= 1) {
    if (tid < off) red[tid] += red[tid + off];
    __syncthreads();
  }
  float simmean = fmaxf(red[0] * (1.0f / BSZ), 0.0f);
  __syncthreads();

  float cfix  = fmaxf(cnt[n], 1.0f);
  float gamma = 1.0f + 1.0f / cfix;
  const float beta = 1.0f + 0.2f * logf(17.0f);   // 1 + beta_alpha*log1p(B)

  float mx = -3.4e38f;
#pragma unroll
  for (int i = 0; i < 16; ++i) {
    v[i] = (v[i] - simmean * beta) * gamma;
    mx = fmaxf(mx, v[i]);
  }
  red[tid] = mx; __syncthreads();
  for (int off = 128; off > 0; off >>= 1) {
    if (tid < off) red[tid] = fmaxf(red[tid], red[tid + off]);
    __syncthreads();
  }
  float rowmax = red[0];
  __syncthreads();
  float dch = fminf(rowmax, CUTV);

  float Z = 0.f;
#pragma unroll
  for (int i = 0; i < 16; ++i) {
    float e = (v[i] >= dch) ? __expf((v[i] - rowmax) * (1.0f / TEMP)) : 0.0f;
    v[i] = e; Z += e;
  }
  red[tid] = Z; __syncthreads();
  for (int off = 128; off > 0; off >>= 1) {
    if (tid < off) red[tid] += red[tid + off];
    __syncthreads();
  }
  float invZ = 1.0f / red[0];
#pragma unroll
  for (int i = 0; i < 16; ++i)
    P[(size_t)n * BSZ + i * 256 + tid] = (half_t)(v[i] * invZ);
}

// ------------------------------------------------------------------
extern "C" void kernel_launch(void* const* d_in, const int* in_sizes, int n_in,
                              void* d_out, int out_size, void* d_ws, size_t ws_size,
                              hipStream_t stream) {
  const float* exf  = (const float*)d_in[0];   // (4,4,768,16,16)
  const float* vext = (const float*)d_in[1];   // (16,12,256,64)
  float* out = (float*)d_out;                  // (256,16,768)

  char* ws = (char*)d_ws;
  size_t off = 0;
  float*  attn   = (float*) (ws + off); off += (size_t)BSZ * BSZ * 4;   // 64 MB
  half_t* P_h    = (half_t*)(ws + off); off += (size_t)BSZ * BSZ * 2;   // 32 MB
  float*  prox32 = (float*) (ws + off); off += (size_t)BSZ * CC * 4;    // 12 MB
  half_t* key_h  = (half_t*)(ws + off); off += (size_t)BSZ * CC * 2;
  half_t* keyT_h = (half_t*)(ws + off); off += (size_t)CC * BSZ * 2;
  half_t* prox_h = (half_t*)(ws + off); off += (size_t)BSZ * CC * 2;
  half_t* VT_h   = (half_t*)(ws + off); off += (size_t)CC * BSZ * 2;
  float*  cnt    = (float*) (ws + off); off += (size_t)BSZ * 4;

  // 0) build normalized key (+transpose) and V^T
  k_build_key<<<BSZ, 256, 0, stream>>>(exf, key_h, keyT_h);
  k_build_v<<<(BSZ * CC) / 256, 256, 0, stream>>>(vext, VT_h);

  // 1) attn = key . key^T
  {
    int tiles = (BSZ / 32) * (BSZ / 32);   // 16384 tiles of 32x32
    k_gemm_nt<<<tiles / 4, 128, 0, stream>>>(key_h, key_h, attn, BSZ, BSZ, CC);
  }

  // 2) two refinement iterations
  for (int it = 0; it < 2; ++it) {
    k_count<<<BSZ, 256, 0, stream>>>(attn, cnt);
    {
      int tiles = (BSZ / 32) * (CC / 32);  // 3072 tiles
      k_gemm_mask<<<tiles / 4, 128, 0, stream>>>(attn, keyT_h, prox32, BSZ, CC, BSZ);
    }
    k_norm_proxy<<<BSZ, 256, 0, stream>>>(prox32, key_h, cnt, prox_h);
    {
      int tiles = (BSZ / 32) * (BSZ / 32);
      k_gemm_nt<<<tiles / 4, 128, 0, stream>>>(prox_h, key_h, attn, BSZ, BSZ, CC);
    }
  }

  // 3) token-norm + dynamic cut + softmax -> f16 P
  k_softmax<<<BSZ, 256, 0, stream>>>(attn, cnt, P_h);

  // 4) out = P . V   (row-permuted write to (S,B,C))
  {
    int tiles = (BSZ / 32) * (CC / 32);
    k_gemm_out<<<tiles / 4, 128, 0, stream>>>(P_h, VT_h, out);
  }
  (void)in_sizes; (void)n_in; (void)out_size; (void)ws_size;
}
